// EncoderAttDecoder_56642028699767
// MI455X (gfx1250) — matrix operand order repaired and checked by reading, not compiled, not run on hardware
//
#include <hip/hip_runtime.h>
#include <hip/hip_bf16.h>

#define L_SEQ   1000
#define M_PAD   1008   // L_SEQ rounded up to 16 (WMMA tile M); padded rows never stored
#define E_DIM   128
#define H_DIM   128
#define D_DIM   256
#define T_TAGS  6
#define START_TAG 4
#define STOP_TAG  5

typedef _Float16 v16h __attribute__((ext_vector_type(16)));
typedef _Float16 v8h  __attribute__((ext_vector_type(8)));
typedef float    v8f  __attribute__((ext_vector_type(8)));

union U16h { v16h v; _Float16 e[16]; v8h h[2]; };
union U8f  { v8f  v; float e[8]; };

__device__ __forceinline__ float sigmoidf_(float x) { return 1.f / (1.f + __expf(-x)); }

// ---------------- embedding gather -> f16 ----------------
__global__ void k_embed(const int* __restrict__ x, const float* __restrict__ emb,
                        _Float16* __restrict__ out) {
    int t = blockIdx.x, i = threadIdx.x;
    out[(size_t)t * E_DIM + i] = (_Float16)emb[(size_t)x[t] * E_DIM + i];
}

// ---------------- f32 -> f16 convert ----------------
__global__ void k_f2h(const float* __restrict__ src, _Float16* __restrict__ dst, int n) {
    int i = blockIdx.x * blockDim.x + threadIdx.x;
    if (i < n) dst[i] = (_Float16)src[i];
}

// ---------------- WMMA GEMM: C[M,N] = A[Mpad,K] * W[N,K]^T + bias[N] ----------------
// One wave per 16x16 output tile; K stepped by 32 with v_wmma_f32_16x16x32_f16.
// A is allocated with Mpad = ceil16(M) rows, so loads are UNGUARDED (a garbage A
// row m only affects D row m, and rows m >= M are masked at the store epilogue).
// CDNA5 wave32 layouts:
//   A (16x32 f16): lane l -> row M=l&15; elem e -> K = (e>>3)*16 + (l>>4)*8 + (e&7)
//   B (32x16 f16): lane l -> col N=l&15; elem e -> K = (e>>3)*16 + (l>>4)*8 + (e&7)
//   D (16x16 f32): vgpr r, lane l -> M = r + 8*(l>>4), N = l&15
__global__ void __launch_bounds__(256)
k_wmma_gemm(const _Float16* __restrict__ A, const _Float16* __restrict__ W,
            const float* __restrict__ bias, float* __restrict__ C,
            int M, int N, int K) {
    int wave   = blockIdx.x * (blockDim.x >> 5) + (threadIdx.x >> 5);
    int tilesN = N >> 4;
    int tilesM = (M + 15) >> 4;
    if (wave >= tilesM * tilesN) return;
    int tM = wave / tilesN, tN = wave % tilesN;
    int lane = threadIdx.x & 31;
    int lrow = lane & 15;          // A-row / B-col / D-col
    int hi   = lane >> 4;          // lane half selects K-group / D-row offset
    int m0 = tM << 4, n0 = tN << 4;

    U8f c;
    #pragma unroll
    for (int i = 0; i < 8; ++i) c.e[i] = 0.f;

    const _Float16* ap0 = A + (size_t)(m0 + lrow) * K + hi * 8;
    const _Float16* bp0 = W + (size_t)(n0 + lrow) * K + hi * 8;

    for (int k0 = 0; k0 < K; k0 += 32) {
        U16h a, b;
        a.h[0] = *reinterpret_cast<const v8h*>(ap0 + k0);
        a.h[1] = *reinterpret_cast<const v8h*>(ap0 + k0 + 16);
        b.h[0] = *reinterpret_cast<const v8h*>(bp0 + k0);
        b.h[1] = *reinterpret_cast<const v8h*>(bp0 + k0 + 16);

        c.v = __builtin_amdgcn_wmma_f32_16x16x32_f16(
            /*neg_a=*/false, a.v, /*neg_b=*/false, b.v,
            /*c_mod=*/(short)0, c.v, /*reuse_a=*/false, /*reuse_b=*/false);
    }

    int n = n0 + lrow;
    float bv = bias[n];
    #pragma unroll
    for (int r = 0; r < 8; ++r) {
        int m = m0 + r + hi * 8;
        if (m < M) C[(size_t)m * N + n] = c.e[r] + bv;
    }
}

// ---------------- encoder layer0 recurrence (2 dirs x 128 units) ----------------
__global__ void __launch_bounds__(256)
k_enc0_scan(const float* __restrict__ GI, const float* __restrict__ Whh,
            const float* __restrict__ bhh, const float* __restrict__ h0,
            _Float16* __restrict__ o16, float* __restrict__ dhid) {
    __shared__ float h[2 * H_DIM];
    int tid = threadIdx.x;
    int d = tid >> 7, u = tid & 127;
    h[tid] = h0[d * H_DIM + u];
    const float* Wr = Whh + ((size_t)d * 3 * H_DIM + u) * H_DIM;
    const float* Wz = Whh + ((size_t)d * 3 * H_DIM + H_DIM + u) * H_DIM;
    const float* Wn = Whh + ((size_t)d * 3 * H_DIM + 2 * H_DIM + u) * H_DIM;
    float br = bhh[d * 3 * H_DIM + u];
    float bz = bhh[d * 3 * H_DIM + H_DIM + u];
    float bn = bhh[d * 3 * H_DIM + 2 * H_DIM + u];
    __syncthreads();
    for (int t = 0; t < L_SEQ; ++t) {
        const float* hh = h + d * H_DIM;
        float ar = 0.f, az = 0.f, an = 0.f;
        #pragma unroll 4
        for (int k = 0; k < H_DIM; ++k) {
            float hv = hh[k];
            ar += Wr[k] * hv; az += Wz[k] * hv; an += Wn[k] * hv;
        }
        const float* gi = GI + (size_t)t * 768 + d * 384;   // bias bih already folded in GEMM
        float r  = sigmoidf_(gi[u] + ar + br);
        float z  = sigmoidf_(gi[H_DIM + u] + az + bz);
        float n  = tanhf(gi[2 * H_DIM + u] + r * (an + bn));
        float hn = (1.f - z) * n + z * h[tid];
        __syncthreads();
        h[tid] = hn;
        o16[(size_t)t * 2 * H_DIM + tid] = (_Float16)hn;
        __syncthreads();
    }
    if (tid < H_DIM) dhid[tid] = h[tid];          // hT[0] = layer0-fwd final
}

// ---------------- encoder layer1 recurrence + per-step log_softmax ----------------
__global__ void __launch_bounds__(256)
k_enc1_scan(const float* __restrict__ GI, const float* __restrict__ Whh,
            const float* __restrict__ bhh, const float* __restrict__ h0,
            float* __restrict__ eouts, float* __restrict__ dhid) {
    __shared__ float h[2 * H_DIM];
    __shared__ float outv[2 * H_DIM];
    __shared__ float red[2 * H_DIM];
    int tid = threadIdx.x;
    int d = tid >> 7, u = tid & 127;
    h[tid] = h0[(2 + d) * H_DIM + u];
    const float* Wr = Whh + ((size_t)d * 3 * H_DIM + u) * H_DIM;
    const float* Wz = Whh + ((size_t)d * 3 * H_DIM + H_DIM + u) * H_DIM;
    const float* Wn = Whh + ((size_t)d * 3 * H_DIM + 2 * H_DIM + u) * H_DIM;
    float br = bhh[d * 3 * H_DIM + u];
    float bz = bhh[d * 3 * H_DIM + H_DIM + u];
    float bn = bhh[d * 3 * H_DIM + 2 * H_DIM + u];
    __syncthreads();
    for (int t = 0; t < L_SEQ; ++t) {
        const float* hh = h + d * H_DIM;
        float ar = 0.f, az = 0.f, an = 0.f;
        #pragma unroll 4
        for (int k = 0; k < H_DIM; ++k) {
            float hv = hh[k];
            ar += Wr[k] * hv; az += Wz[k] * hv; an += Wn[k] * hv;
        }
        const float* gi = GI + (size_t)t * 768 + d * 384;
        float r  = sigmoidf_(gi[u] + ar + br);
        float z  = sigmoidf_(gi[H_DIM + u] + az + bz);
        float n  = tanhf(gi[2 * H_DIM + u] + r * (an + bn));
        float hn = (1.f - z) * n + z * h[tid];
        __syncthreads();                 // all dot-products done before h update
        h[tid] = hn;
        outv[tid] = hn;
        __syncthreads();
        // log_softmax over the 256-vector concat(h1f, h1b)
        red[tid] = outv[tid];
        __syncthreads();
        for (int s2 = 128; s2 > 0; s2 >>= 1) {
            if (tid < s2) red[tid] = fmaxf(red[tid], red[tid + s2]);
            __syncthreads();
        }
        float mx = red[0];
        __syncthreads();
        red[tid] = __expf(outv[tid] - mx);
        __syncthreads();
        for (int s2 = 128; s2 > 0; s2 >>= 1) {
            if (tid < s2) red[tid] += red[tid + s2];
            __syncthreads();
        }
        float lse = mx + __logf(red[0]);
        eouts[(size_t)t * 2 * H_DIM + tid] = outv[tid] - lse;
        __syncthreads();
    }
    if (tid >= H_DIM) dhid[tid] = h[tid];         // hT[3] = layer1-bwd final
}

// ---------------- decoder step for one candidate tag (hidden is static) ----------------
__global__ void __launch_bounds__(256)
k_decoder(const float* __restrict__ de_embed, const float* __restrict__ attn_W,
          const float* __restrict__ attn_b, const float* __restrict__ comb_W,
          const float* __restrict__ comb_b, const float* __restrict__ de_Wih,
          const float* __restrict__ de_Whh, const float* __restrict__ de_bih,
          const float* __restrict__ de_bhh, const float* __restrict__ h2t_W,
          const float* __restrict__ h2t_b, const float* __restrict__ eouts,
          const float* __restrict__ dhid, int* __restrict__ fmap,
          float* __restrict__ gvals) {
    __shared__ float aw[L_SEQ];
    __shared__ float cat1[2 * D_DIM];   // [emb, dhid]
    __shared__ float cat2[2 * D_DIM];   // [emb, app]
    __shared__ float oact[D_DIM];
    __shared__ float hnS[D_DIM];
    __shared__ float red[256];
    __shared__ float lg[T_TAGS];
    int din = blockIdx.x, tid = threadIdx.x;

    cat1[tid] = de_embed[(size_t)din * D_DIM + tid];
    cat1[D_DIM + tid] = dhid[tid];
    cat2[tid] = cat1[tid];
    __syncthreads();

    // attention logits
    for (int t = tid; t < L_SEQ; t += 256) {
        const float* w = attn_W + (size_t)t * 2 * D_DIM;
        float acc = attn_b[t];
        for (int j = 0; j < 2 * D_DIM; ++j) acc += w[j] * cat1[j];
        aw[t] = acc;
    }
    __syncthreads();
    // softmax over L
    float lm = -3.0e38f;
    for (int t = tid; t < L_SEQ; t += 256) lm = fmaxf(lm, aw[t]);
    red[tid] = lm; __syncthreads();
    for (int s2 = 128; s2 > 0; s2 >>= 1) {
        if (tid < s2) red[tid] = fmaxf(red[tid], red[tid + s2]);
        __syncthreads();
    }
    float mx = red[0]; __syncthreads();
    float ls = 0.f;
    for (int t = tid; t < L_SEQ; t += 256) { float e = __expf(aw[t] - mx); aw[t] = e; ls += e; }
    red[tid] = ls; __syncthreads();
    for (int s2 = 128; s2 > 0; s2 >>= 1) {
        if (tid < s2) red[tid] += red[tid + s2];
        __syncthreads();
    }
    float sum = red[0];
    __syncthreads();
    // app = aw @ eouts (column per thread, coalesced over lanes)
    {
        float acc = 0.f;
        for (int t = 0; t < L_SEQ; ++t) acc += aw[t] * eouts[(size_t)t * D_DIM + tid];
        cat2[D_DIM + tid] = acc / sum;
    }
    __syncthreads();
    // o = relu(comb_W @ cat2 + comb_b)
    {
        const float* w = comb_W + (size_t)tid * 2 * D_DIM;
        float a2 = comb_b[tid];
        for (int j = 0; j < 2 * D_DIM; ++j) a2 += w[j] * cat2[j];
        oact[tid] = fmaxf(a2, 0.f);
    }
    __syncthreads();
    // GRU cell: x = oact, h = dhid (cat1 upper half)
    {
        const float* Wir = de_Wih + (size_t)tid * D_DIM;
        const float* Wiz = de_Wih + (size_t)(D_DIM + tid) * D_DIM;
        const float* Win = de_Wih + (size_t)(2 * D_DIM + tid) * D_DIM;
        const float* Whr = de_Whh + (size_t)tid * D_DIM;
        const float* Whz = de_Whh + (size_t)(D_DIM + tid) * D_DIM;
        const float* Whn = de_Whh + (size_t)(2 * D_DIM + tid) * D_DIM;
        float ir = de_bih[tid], iz = de_bih[D_DIM + tid], iN = de_bih[2 * D_DIM + tid];
        float hr = de_bhh[tid], hz = de_bhh[D_DIM + tid], hN = de_bhh[2 * D_DIM + tid];
        for (int j = 0; j < D_DIM; ++j) {
            float xv = oact[j], hv = cat1[D_DIM + j];
            ir += Wir[j] * xv; iz += Wiz[j] * xv; iN += Win[j] * xv;
            hr += Whr[j] * hv; hz += Whz[j] * hv; hN += Whn[j] * hv;
        }
        float r = sigmoidf_(ir + hr);
        float z = sigmoidf_(iz + hz);
        float n = tanhf(iN + r * hN);
        hnS[tid] = (1.f - z) * n + z * cat1[D_DIM + tid];
    }
    __syncthreads();
    if (tid < T_TAGS) {
        const float* w = h2t_W + (size_t)tid * D_DIM;
        float a2 = h2t_b[tid];
        for (int j = 0; j < D_DIM; ++j) a2 += w[j] * hnS[j];
        lg[tid] = a2;
    }
    __syncthreads();
    if (tid == 0) {
        float m2 = lg[0]; int bi = 0;
        for (int k = 1; k < T_TAGS; ++k) if (lg[k] > m2) { m2 = lg[k]; bi = k; }
        float s2 = 0.f;
        for (int k = 0; k < T_TAGS; ++k) s2 += __expf(lg[k] - m2);
        fmap[din]  = bi;                         // next token
        gvals[din] = lg[0] - (m2 + __logf(s2));  // logp[0]
    }
}

// ---------------- tag chain + rank-1 feats + Viterbi + backtrack ----------------
__global__ void __launch_bounds__(64)
k_chain_vit(const float* __restrict__ h2t_W, const float* __restrict__ h2t_b,
            const float* __restrict__ trans, const int* __restrict__ fmap,
            const float* __restrict__ gvals, int* __restrict__ bptrs,
            float* __restrict__ out) {
    __shared__ float s[L_SEQ];
    __shared__ float rs[T_TAGS], tb[T_TAGS], fv[T_TAGS], nfv[T_TAGS];
    int tid = threadIdx.x;
    if (tid < T_TAGS) {
        float a = 0.f;
        for (int j = 0; j < D_DIM; ++j) a += h2t_W[(size_t)tid * D_DIM + j];
        rs[tid] = a; tb[tid] = h2t_b[tid];
        fv[tid] = (tid == START_TAG) ? 0.f : -10000.f;
    }
    if (tid == 0) {
        int din = START_TAG;
        for (int t = 0; t < L_SEQ; ++t) { s[t] = gvals[din]; din = fmap[din]; }
    }
    __syncthreads();
    for (int t = 0; t < L_SEQ; ++t) {
        if (tid < T_TAGS) {
            float best = fv[0] + trans[tid * T_TAGS];
            int bj = 0;
            for (int j = 1; j < T_TAGS; ++j) {
                float v = fv[j] + trans[tid * T_TAGS + j];
                if (v > best) { best = v; bj = j; }
            }
            nfv[tid] = best + s[t] * rs[tid] + tb[tid];
            bptrs[t * T_TAGS + tid] = bj;
        }
        __syncthreads();
        if (tid < T_TAGS) fv[tid] = nfv[tid];
        __syncthreads();
    }
    if (tid == 0) {
        float best = fv[0] + trans[STOP_TAG * T_TAGS];
        int bi = 0;
        for (int k = 1; k < T_TAGS; ++k) {
            float v = fv[k] + trans[STOP_TAG * T_TAGS + k];
            if (v > best) { best = v; bi = k; }
        }
        out[L_SEQ - 1] = (float)bi;
        int cur = bi;
        for (int j = L_SEQ - 2; j >= 0; --j) {
            cur = bptrs[(j + 1) * T_TAGS + cur];
            out[j] = (float)cur;
        }
    }
}

extern "C" void kernel_launch(void* const* d_in, const int* in_sizes, int n_in,
                              void* d_out, int out_size, void* d_ws, size_t ws_size,
                              hipStream_t stream) {
    (void)in_sizes; (void)n_in; (void)out_size; (void)ws_size;
    const int*   x        = (const int*)  d_in[0];
    const float* en_embed = (const float*)d_in[1];
    const float* e0Wih    = (const float*)d_in[2];
    const float* e0Whh    = (const float*)d_in[3];
    const float* e0bih    = (const float*)d_in[4];
    const float* e0bhh    = (const float*)d_in[5];
    const float* e1Wih    = (const float*)d_in[6];
    const float* e1Whh    = (const float*)d_in[7];
    const float* e1bih    = (const float*)d_in[8];
    const float* e1bhh    = (const float*)d_in[9];
    const float* de_embed = (const float*)d_in[10];
    const float* attn_W   = (const float*)d_in[11];
    const float* attn_b   = (const float*)d_in[12];
    const float* comb_W   = (const float*)d_in[13];
    const float* comb_b   = (const float*)d_in[14];
    const float* de_Wih   = (const float*)d_in[15];
    const float* de_Whh   = (const float*)d_in[16];
    const float* de_bih   = (const float*)d_in[17];
    const float* de_bhh   = (const float*)d_in[18];
    const float* h2t_W    = (const float*)d_in[19];
    const float* h2t_b    = (const float*)d_in[20];
    const float* trans    = (const float*)d_in[21];
    const float* h0       = (const float*)d_in[22];

    char* base = (char*)d_ws; size_t off = 0;
    auto alloc = [&](size_t b) -> void* {
        void* p = base + off;
        off += (b + 255) & ~(size_t)255;
        return p;
    };
    // A operands padded to M_PAD rows so WMMA loads need no bounds checks.
    _Float16* embs16 = (_Float16*)alloc((size_t)M_PAD * E_DIM * 2);
    _Float16* w016   = (_Float16*)alloc((size_t)768 * 128 * 2);
    _Float16* w116   = (_Float16*)alloc((size_t)768 * 256 * 2);
    float*    GI0    = (float*)   alloc((size_t)L_SEQ * 768 * 4);
    float*    GI1    = (float*)   alloc((size_t)L_SEQ * 768 * 4);
    _Float16* o016   = (_Float16*)alloc((size_t)M_PAD * 256 * 2);
    float*    eouts  = (float*)   alloc((size_t)L_SEQ * 256 * 4);
    float*    dhid   = (float*)   alloc((size_t)256 * 4);
    int*      fmap   = (int*)     alloc((size_t)T_TAGS * 4);
    float*    gvals  = (float*)   alloc((size_t)T_TAGS * 4);
    int*      bptrs  = (int*)     alloc((size_t)L_SEQ * T_TAGS * 4);

    // Phase A: gather embeddings + f16 weight copies
    k_embed<<<L_SEQ, E_DIM, 0, stream>>>(x, en_embed, embs16);
    k_f2h<<<(768 * 128 + 255) / 256, 256, 0, stream>>>(e0Wih, w016, 768 * 128);
    k_f2h<<<(768 * 256 + 255) / 256, 256, 0, stream>>>(e1Wih, w116, 768 * 256);

    // Phase B: batched input projections via WMMA (time-parallel part of layer 0, both directions)
    int tilesM = (L_SEQ + 15) / 16;          // 63
    int tilesN = 768 / 16;                   // 48
    int blocks = (tilesM * tilesN + 7) / 8;  // 8 waves per 256-thread block
    k_wmma_gemm<<<blocks, 256, 0, stream>>>(embs16, w016, e0bih, GI0, L_SEQ, 768, 128);

    // Phase C: layer-0 recurrence (produces o0 for layer-1's input GEMM)
    k_enc0_scan<<<1, 256, 0, stream>>>(GI0, e0Whh, e0bhh, h0, o016, dhid);

    // Phase D: layer-1 input projection via WMMA, then layer-1 recurrence + log_softmax
    k_wmma_gemm<<<blocks, 256, 0, stream>>>(o016, w116, e1bih, GI1, L_SEQ, 768, 256);
    k_enc1_scan<<<1, 256, 0, stream>>>(GI1, e1Whh, e1bhh, h0, eouts, dhid);

    // Phase E: decoder transition function over its 6-element domain (hidden never updates)
    k_decoder<<<T_TAGS, 256, 0, stream>>>(de_embed, attn_W, attn_b, comb_W, comb_b,
                                          de_Wih, de_Whh, de_bih, de_bhh,
                                          h2t_W, h2t_b, eouts, dhid, fmap, gvals);

    // Phase F: 1000-step tag chain + rank-1 feats + Viterbi + backtrack
    k_chain_vit<<<1, 64, 0, stream>>>(h2t_W, h2t_b, trans, fmap, gvals, bptrs,
                                      (float*)d_out);
}
